// PattentionLayer_47055661695400
// MI455X (gfx1250) — matrix operand order repaired and checked by reading, not compiled, not run on hardware
//
#include <hip/hip_runtime.h>
#include <hip/hip_bf16.h>

typedef _Float16 h16;
typedef __attribute__((ext_vector_type(16))) _Float16 v16h;
typedef __attribute__((ext_vector_type(8)))  _Float16 v8h;
typedef __attribute__((ext_vector_type(4)))  _Float16 v4h;
typedef __attribute__((ext_vector_type(8)))  float    v8f;
typedef __attribute__((ext_vector_type(4)))  float    v4f;

#define DIN    512
#define DOUT   512
#define NPT    2048
#define MTOT   16384   // B*S = 4*4096
#define MTILE  32      // rows per workgroup (2 row-blocks of 16)
#define NTHR   512     // 16 waves (wave32)

// ---------- prepass 1: K (NPT x DIN fp32) -> f16, same layout ----------
__global__ void cvt_f16_kernel(const float* __restrict__ src, h16* __restrict__ dst, int n4) {
  int i = blockIdx.x * blockDim.x + threadIdx.x;
  if (i < n4) {
    v4f v = ((const v4f*)src)[i];
    v4h o = { (h16)v.x, (h16)v.y, (h16)v.z, (h16)v.w };
    ((v4h*)dst)[i] = o;
  }
}

// ---------- prepass 2: V (NPT x DOUT fp32) -> Vt (DOUT x NPT f16) ----------
__global__ void transpose_v_kernel(const float* __restrict__ V, h16* __restrict__ Vt) {
  __shared__ float tile[32][33];
  int n0 = blockIdx.x * 32;
  int d0 = blockIdx.y * 32;
  int tx = threadIdx.x, ty = threadIdx.y;   // block (32, 8)
  #pragma unroll
  for (int j = 0; j < 32; j += 8)
    tile[ty + j][tx] = V[(size_t)(n0 + ty + j) * DOUT + (d0 + tx)];
  __syncthreads();
  #pragma unroll
  for (int j = 0; j < 32; j += 8)
    Vt[(size_t)(d0 + ty + j) * NPT + (n0 + tx)] = (h16)tile[tx][ty + j];
}

// Load one 16x32 f16 A-tile from row-major f16 LDS in contiguous-k order (k = 16*hi + s; B loads use the same):
//   lanes 0-15 : row = base+lane,    halves 0-15 <- K[k0..k0+15]
//   lanes16-31 : row = base+lane-16, halves 0-15 <- K[k0+16..k0+31]
__device__ __forceinline__ v16h load_a_tile(const h16* lds, int W, int row, int k0, int hi) {
  const h16* p = lds + row * W + k0 + hi * 16;
  v8h lo = *(const v8h*)(p);
  v8h hh = *(const v8h*)(p + 8);
  v16h a;
  #pragma unroll
  for (int i = 0; i < 8; ++i) { a[i] = lo[i]; a[8 + i] = hh[i]; }
  return a;
}

__launch_bounds__(NTHR)
__global__ void pattention_fused_kernel(const float* __restrict__ X,
                                        const h16*  __restrict__ Kf,   // NPT x DIN, f16
                                        const h16*  __restrict__ Vt,   // DOUT x NPT, f16
                                        float*      __restrict__ Out) {
  __shared__ h16   ldsX[MTILE * DIN];   //  32 KB: x tile, f16 row-major
  __shared__ h16   ldsA[MTILE * NPT];   // 128 KB: gelu(norm(sim)) tile, f16 row-major
  __shared__ float sumsq[MTILE];

  const int tid  = threadIdx.x;
  const int wave = tid >> 5;            // 16 waves
  const int lane = tid & 31;
  const int colL = lane & 15;
  const int hi   = lane >> 4;
  const size_t m0 = (size_t)blockIdx.x * MTILE;

  // ---- stage 32x512 x-tile -> LDS f16 (rows contiguous in memory) ----
  {
    const v4f* xs = (const v4f*)(X + m0 * DIN);
    v4h* xd = (v4h*)ldsX;
    #pragma unroll
    for (int it = 0; it < (MTILE * DIN / 4) / NTHR; ++it) {
      int i = it * NTHR + tid;
      v4f v = xs[i];
      v4h o = { (h16)v.x, (h16)v.y, (h16)v.z, (h16)v.w };
      xd[i] = o;
    }
  }
  if (tid < MTILE) sumsq[tid] = 0.0f;
  __syncthreads();

  const v8f vzero = {0.f,0.f,0.f,0.f,0.f,0.f,0.f,0.f};

  // ---- GEMM1: sim[32][2048] = X @ K^T ----
  // wave owns n in [wave*128, wave*128+128): 8 n-tiles x 2 row-blocks = 16 accs.
  // Each B tile is reused by 2 WMMAs (both row-blocks) -> 32 FLOP per L2 byte.
  v8f acc[2][8];
  #pragma unroll
  for (int rb = 0; rb < 2; ++rb)
    #pragma unroll
    for (int t = 0; t < 8; ++t) acc[rb][t] = vzero;

  const int n_base = wave * 128;
  for (int k0 = 0; k0 < DIN; k0 += 32) {
    v16h a0 = load_a_tile(ldsX, DIN, colL,      k0, hi);
    v16h a1 = load_a_tile(ldsX, DIN, 16 + colL, k0, hi);
    const h16* bp = Kf + (size_t)(n_base + colL) * DIN + k0 + hi * 16;
    // 4-deep rotating B pipeline: keeps >=4 tiles of loads in flight so the
    // scheduler can wait with slack instead of s_wait_loadcnt 0 per WMMA.
    v16h bb[4];
    #pragma unroll
    for (int j = 0; j < 4; ++j) bb[j] = *(const v16h*)(bp + (size_t)j * 16 * DIN);
    #pragma unroll
    for (int t = 0; t < 8; ++t) {
      v16h bc = bb[t & 3];
      if (t + 4 < 8) bb[t & 3] = *(const v16h*)(bp + (size_t)(t + 4) * 16 * DIN);
      acc[0][t] = __builtin_amdgcn_wmma_f32_16x16x32_f16(false, a0, false, bc,
                                                         (short)0, acc[0][t], false, false);
      acc[1][t] = __builtin_amdgcn_wmma_f32_16x16x32_f16(false, a1, false, bc,
                                                         (short)0, acc[1][t], false, false);
    }
  }

  // ---- per-row sum of squares (C layout: lane holds 8 rows x 1 col per tile) ----
  float p[2][8];
  #pragma unroll
  for (int rb = 0; rb < 2; ++rb)
    #pragma unroll
    for (int r = 0; r < 8; ++r) p[rb][r] = 0.0f;
  #pragma unroll
  for (int rb = 0; rb < 2; ++rb)
    #pragma unroll
    for (int t = 0; t < 8; ++t)
      #pragma unroll
      for (int r = 0; r < 8; ++r) p[rb][r] += acc[rb][t][r] * acc[rb][t][r];
  #pragma unroll
  for (int m = 8; m >= 1; m >>= 1)
    #pragma unroll
    for (int rb = 0; rb < 2; ++rb)
      #pragma unroll
      for (int r = 0; r < 8; ++r) p[rb][r] += __shfl_xor(p[rb][r], m, 32);
  if (colL == 0) {                       // lanes 0 / 16 hold rows 0-7 / 8-15 of each block
    #pragma unroll
    for (int rb = 0; rb < 2; ++rb)
      #pragma unroll
      for (int r = 0; r < 8; ++r)
        atomicAdd(&sumsq[rb * 16 + hi * 8 + r], p[rb][r]);
  }
  __syncthreads();

  float inv[2][8];
  #pragma unroll
  for (int rb = 0; rb < 2; ++rb)
    #pragma unroll
    for (int r = 0; r < 8; ++r) {
      float nrm = sqrtf(sumsq[rb * 16 + hi * 8 + r]);
      inv[rb][r] = 1.0f / fmaxf(nrm, 1e-12f);   // d^-1/2 scale cancels under L2-normalize
    }

  // ---- normalize + exact GELU -> ldsA (f16, row-major 32 x 2048) ----
  #pragma unroll
  for (int rb = 0; rb < 2; ++rb)
    #pragma unroll
    for (int t = 0; t < 8; ++t) {
      int col = n_base + t * 16 + colL;
      #pragma unroll
      for (int r = 0; r < 8; ++r) {
        float v = acc[rb][t][r] * inv[rb][r];
        float g = 0.5f * v * (1.0f + erff(v * 0.70710678118654752f));
        ldsA[(rb * 16 + hi * 8 + r) * NPT + col] = (h16)g;
      }
    }
  __syncthreads();

  // ---- GEMM2: out[32][512] = attn @ V ; wave owns d in [wave*32, wave*32+32) ----
  // 2 n-tiles x 2 row-blocks; A/B double-buffered across k (short body: 4 WMMAs).
  v8f oacc[2][2];
  #pragma unroll
  for (int rb = 0; rb < 2; ++rb)
    #pragma unroll
    for (int u = 0; u < 2; ++u) oacc[rb][u] = vzero;

  const int d_base = wave * 32;
  const h16* vb = Vt + (size_t)(d_base + colL) * NPT + hi * 16;

  v16h a_c[2], b_c[2];
  #pragma unroll
  for (int rb = 0; rb < 2; ++rb) a_c[rb] = load_a_tile(ldsA, NPT, rb * 16 + colL, 0, hi);
  #pragma unroll
  for (int u = 0; u < 2; ++u) b_c[u] = *(const v16h*)(vb + (size_t)u * 16 * NPT);

  for (int k0 = 0; k0 < NPT; k0 += 32) {
    v16h a_n[2], b_n[2];
    const int k1 = k0 + 32;
    if (k1 < NPT) {
      #pragma unroll
      for (int rb = 0; rb < 2; ++rb) a_n[rb] = load_a_tile(ldsA, NPT, rb * 16 + colL, k1, hi);
      #pragma unroll
      for (int u = 0; u < 2; ++u) b_n[u] = *(const v16h*)(vb + k1 + (size_t)u * 16 * NPT);
    } else {
      #pragma unroll
      for (int rb = 0; rb < 2; ++rb) a_n[rb] = a_c[rb];
      #pragma unroll
      for (int u = 0; u < 2; ++u)  b_n[u] = b_c[u];
    }
    #pragma unroll
    for (int u = 0; u < 2; ++u)
      #pragma unroll
      for (int rb = 0; rb < 2; ++rb)
        oacc[rb][u] = __builtin_amdgcn_wmma_f32_16x16x32_f16(false, a_c[rb], false, b_c[u],
                                                             (short)0, oacc[rb][u], false, false);
    #pragma unroll
    for (int rb = 0; rb < 2; ++rb) a_c[rb] = a_n[rb];
    #pragma unroll
    for (int u = 0; u < 2; ++u)  b_c[u] = b_n[u];
  }

  // ---- store fp32 output (C layout -> row-major global) ----
  #pragma unroll
  for (int rb = 0; rb < 2; ++rb)
    #pragma unroll
    for (int u = 0; u < 2; ++u) {
      int col = d_base + u * 16 + colL;
      #pragma unroll
      for (int r = 0; r < 8; ++r)
        Out[(m0 + rb * 16 + hi * 8 + r) * DOUT + col] = oacc[rb][u][r];
    }
}

extern "C" void kernel_launch(void* const* d_in, const int* in_sizes, int n_in,
                              void* d_out, int out_size, void* d_ws, size_t ws_size,
                              hipStream_t stream) {
  (void)in_sizes; (void)n_in; (void)out_size; (void)ws_size;
  const float* X  = (const float*)d_in[0];   // (4,4096,512)  fp32
  const float* Kp = (const float*)d_in[1];   // (2048,512)    fp32
  const float* Vp = (const float*)d_in[2];   // (2048,512)    fp32
  float* Out = (float*)d_out;                // (4,4096,512)  fp32

  h16* Kf = (h16*)d_ws;                                                  // 2 MB
  h16* Vt = (h16*)((char*)d_ws + (size_t)NPT * DIN * sizeof(h16));       // 2 MB

  {
    int n4 = NPT * DIN / 4;
    cvt_f16_kernel<<<(n4 + 255) / 256, 256, 0, stream>>>(Kp, Kf, n4);
  }
  {
    dim3 g(NPT / 32, DOUT / 32), b(32, 8);
    transpose_v_kernel<<<g, b, 0, stream>>>(Vp, Vt);
  }
  pattention_fused_kernel<<<MTOT / MTILE, NTHR, 0, stream>>>(X, Kf, Vt, Out);
}
